// CmhAttCPI_52587579572419
// MI455X (gfx1250) — compile-verified, matrix operand order checked by reading, not run on hardware
//
#include <hip/hip_runtime.h>
#include <cstddef>
#include <cstdint>

// ---------------------------------------------------------------------------
// Types for CDNA5 WMMA (wave32).
// ---------------------------------------------------------------------------
typedef __bf16 bf16;
typedef __attribute__((ext_vector_type(16))) __bf16 v16bf;
typedef __attribute__((ext_vector_type(8)))  __bf16 v8bf;
typedef __attribute__((ext_vector_type(8)))  float  v8f;

#define DEV __device__ __forceinline__

// Load one 16-lane-row fragment chunk: 8 bf16 at p, 8 bf16 at p+16 (matches the
// ISA 16-bit A/B layout: lane group g holds K = 8g..8g+7 and 16+8g..16+8g+7).
DEV v16bf load_frag(const bf16* p) {
  v8bf lo = *(const v8bf*)p;
  v8bf hi = *(const v8bf*)(p + 16);
  v16bf r;
#pragma unroll
  for (int i = 0; i < 8; ++i) { r[i] = lo[i]; r[i + 8] = hi[i]; }
  return r;
}

DEV float sigmoidf_(float x) { return 1.f / (1.f + __expf(-x)); }
DEV float lrelu_(float x)    { return x >= 0.f ? x : 0.1f * x; }

// ---------------------------------------------------------------------------
// Generic bf16 WMMA GEMM: C[M,Ntot] = act( A[M,K] * Bt[Ntot,K]^T + bias + rowbias )
// One wave -> 16(M) x 64(N) tile. A row-major (lda), Bt row-major (ldb, row = out col).
// epi: 0=identity 1=lrelu 2=tanh 3=sigmoid.
// Optional fp32 out C and bf16 out Cbf (at row*ldcb + cbf_off + col).
// rowbias (only when Ntot==128): x += rowbias[(row>>rb_shift)*128 + col].
// ---------------------------------------------------------------------------
__global__ void k_gemm(const bf16* __restrict__ A, int lda,
                       const bf16* __restrict__ Bt, int ldb,
                       float* __restrict__ C, bf16* __restrict__ Cbf,
                       int ldcb, int cbf_off,
                       const float* __restrict__ bias,
                       const float* __restrict__ rowbias, int rb_shift,
                       int M, int Ntot, int K, int epi) {
  int wid  = (int)((blockIdx.x * blockDim.x + threadIdx.x) >> 5);
  int lane = (int)(threadIdx.x & 31);
  int ngrps = Ntot >> 6;
  int mt = wid / ngrps;
  int ng = wid - mt * ngrps;
  if (mt >= (M >> 4)) return;
  int n0 = ng << 6;
  int g  = lane >> 4;       // lane half
  int ln = lane & 15;

  const bf16* arow = A  + (size_t)((mt << 4) + ln) * lda + (g << 3);
  const bf16* brow = Bt + (size_t)(n0 + ln) * ldb + (g << 3);

  v8f c0 = {}, c1 = {}, c2 = {}, c3 = {};
  for (int k0 = 0; k0 < K; k0 += 32) {
    v16bf a  = load_frag(arow + k0);
    v16bf b0 = load_frag(brow + k0);
    v16bf b1 = load_frag(brow + (size_t)16 * ldb + k0);
    v16bf b2 = load_frag(brow + (size_t)32 * ldb + k0);
    v16bf b3 = load_frag(brow + (size_t)48 * ldb + k0);
    c0 = __builtin_amdgcn_wmma_f32_16x16x32_bf16(false, a, false, b0, (short)0, c0, false, false);
    c1 = __builtin_amdgcn_wmma_f32_16x16x32_bf16(false, a, false, b1, (short)0, c1, false, false);
    c2 = __builtin_amdgcn_wmma_f32_16x16x32_bf16(false, a, false, b2, (short)0, c2, false, false);
    c3 = __builtin_amdgcn_wmma_f32_16x16x32_bf16(false, a, false, b3, (short)0, c3, false, false);
  }
  v8f cc[4] = {c0, c1, c2, c3};
  int rbase = (mt << 4) + (g << 3);   // C layout: lane half g holds rows rbase..rbase+7
#pragma unroll
  for (int t = 0; t < 4; ++t) {
    int col = n0 + (t << 4) + ln;
    float bv = bias ? bias[col] : 0.f;
#pragma unroll
    for (int v = 0; v < 8; ++v) {
      int row = rbase + v;
      float x = cc[t][v] + bv;
      if (rowbias) x += rowbias[((row >> rb_shift) << 7) + col];
      if (epi == 1)      x = lrelu_(x);
      else if (epi == 2) x = tanhf(x);
      else if (epi == 3) x = sigmoidf_(x);
      if (C)   C[(size_t)row * Ntot + col] = x;
      if (Cbf) Cbf[(size_t)row * ldcb + cbf_off + col] = (bf16)x;
    }
  }
}

// ---------------------------------------------------------------------------
// Fused WLN neighbor GEMM:
//   rows r in [0, B*N*NB): A[r] = [ vf_bf16[aa[r]] (128) | ebf[bb[r]] (6 + pad) ]
//   nei_label = lrelu(A * Wu2T^T + bu2) * nbs_mask, summed over NB=8 rows.
// One wave: 16-row tile x 32 cols; the 8-row neighbor groups land exactly in
// the lane halves of the C fragment, so the NB sum is a per-lane sum over v.
// ---------------------------------------------------------------------------
__global__ void k_nei(const bf16* __restrict__ catbuf,  // [65536,256], vf in cols 0..127
                      const bf16* __restrict__ ebf,     // [131072,32], cols 6..31 zero
                      const int* __restrict__ aa, const int* __restrict__ bb,
                      const bf16* __restrict__ Wu2T,    // [128,160], cols 134..159 zero
                      const float* __restrict__ bu2,
                      const float* __restrict__ nbs_mask,
                      float* __restrict__ nei,          // [65536,128]
                      bf16* __restrict__ catout) {      // cat second half (ld 256)
  int wid  = (int)((blockIdx.x * blockDim.x + threadIdx.x) >> 5);
  int lane = (int)(threadIdx.x & 31);
  int mt = wid >> 2;          // 16-row tile over 524288 rows
  int np = wid & 3;
  int n0 = np << 5;
  int g = lane >> 4, ln = lane & 15;

  int r  = (mt << 4) + ln;
  int ia = aa[r];
  int ib = bb[r];
  const bf16* arow = catbuf + (size_t)ia * 256 + (g << 3);
  const bf16* erow = ebf    + (size_t)ib * 32  + (g << 3);
  const bf16* brow = Wu2T   + (size_t)(n0 + ln) * 160 + (g << 3);

  v8f c0 = {}, c1 = {};
#pragma unroll
  for (int kc = 0; kc < 4; ++kc) {
    int k0 = kc << 5;
    v16bf a  = load_frag(arow + k0);
    v16bf b0 = load_frag(brow + k0);
    v16bf b1 = load_frag(brow + (size_t)16 * 160 + k0);
    c0 = __builtin_amdgcn_wmma_f32_16x16x32_bf16(false, a, false, b0, (short)0, c0, false, false);
    c1 = __builtin_amdgcn_wmma_f32_16x16x32_bf16(false, a, false, b1, (short)0, c1, false, false);
  }
  { // edge-feature K-chunk (k = 128..159, zero padded on both operands)
    v16bf a  = load_frag(erow);
    v16bf b0 = load_frag(brow + 128);
    v16bf b1 = load_frag(brow + (size_t)16 * 160 + 128);
    c0 = __builtin_amdgcn_wmma_f32_16x16x32_bf16(false, a, false, b0, (short)0, c0, false, false);
    c1 = __builtin_amdgcn_wmma_f32_16x16x32_bf16(false, a, false, b1, (short)0, c1, false, false);
  }
  int col0 = n0 + ln, col1 = n0 + 16 + ln;
  float b0v = bu2[col0], b1v = bu2[col1];
  float acc0 = 0.f, acc1 = 0.f;
  int rbase = (mt << 4) + (g << 3);
#pragma unroll
  for (int v = 0; v < 8; ++v) {
    float mk = nbs_mask[rbase + v];
    acc0 += lrelu_(c0[v] + b0v) * mk;
    acc1 += lrelu_(c1[v] + b1v) * mk;
  }
  int orow = (mt << 1) + g;  // (b,n) index
  nei[(size_t)orow * 128 + col0] = acc0;
  nei[(size_t)orow * 128 + col1] = acc1;
  catout[(size_t)orow * 256 + 128 + col0] = (bf16)acc0;
  catout[(size_t)orow * 256 + 128 + col1] = (bf16)acc1;
}

// ---------------------------------------------------------------------------
// Prep / elementwise kernels
// ---------------------------------------------------------------------------
__global__ void k_transpose_pad(const float* __restrict__ in, bf16* __restrict__ out,
                                int Kdim, int Nn, int ldo) {
  int i = (int)(blockIdx.x * blockDim.x + threadIdx.x);
  int tot = Nn * ldo;
  if (i >= tot) return;
  int n = i / ldo, k = i - n * ldo;
  out[i] = (k < Kdim) ? (bf16)in[(size_t)k * Nn + n] : (bf16)0.f;
}

__global__ void k_copy_bf16(const float* __restrict__ in, bf16* __restrict__ out, int n) {
  int i = (int)(blockIdx.x * blockDim.x + threadIdx.x);
  if (i < n) out[i] = (bf16)in[i];
}

__global__ void k_gather_pad(const float* __restrict__ table, int tcols,
                             const int* __restrict__ idx, bf16* __restrict__ out,
                             int ldo, int validc, int nrows) {
  int i = (int)(blockIdx.x * blockDim.x + threadIdx.x);
  int tot = nrows * ldo;
  if (i >= tot) return;
  int r = i / ldo, c = i - r * ldo;
  out[i] = (c < validc) ? (bf16)table[(size_t)idx[r] * tcols + c] : (bf16)0.f;
}

__global__ void k_addvec(const float* a, const float* b, float* o, int n) {
  int i = (int)(blockIdx.x * blockDim.x + threadIdx.x);
  if (i < n) o[i] = a[i] + b[i];
}

__global__ void k_sf_init(const float* __restrict__ vf, const float* __restrict__ vmask,
                          float* __restrict__ sf, bf16* __restrict__ sfbf) {
  int b = blockIdx.x, g = threadIdx.x;
  float acc = 0.f;
  for (int n = 0; n < 256; ++n)
    acc += vf[((size_t)b * 256 + n) * 128 + g] * vmask[b * 256 + n];
  sf[b * 128 + g] = acc;
  sfbf[b * 128 + g] = (bf16)acc;
}

__global__ void k_attn_score(const float* __restrict__ amain, const float* __restrict__ sf,
                             const float* __restrict__ Wbmm_d, const float* __restrict__ bbmm_d,
                             float* __restrict__ score) {
  int t = (int)(blockIdx.x * blockDim.x + threadIdx.x);
  if (t >= 256 * 4 * 256) return;
  int n = t & 255, k = (t >> 8) & 3, b = t >> 10;
  const float* ar = amain + ((size_t)b * 256 + n) * 512 + k * 128;
  const float* sr = sf + (size_t)b * 128;
  const float* wr = Wbmm_d + k * 128;
  float acc = 0.f;
  for (int g2 = 0; g2 < 128; ++g2) acc += ar[g2] * sr[g2] * wr[g2];
  score[((size_t)b * 4 + k) * 256 + n] = acc + bbmm_d[k];
}

__global__ void k_softmax(float* __restrict__ score, const float* __restrict__ vmask) {
  __shared__ float sh[256];
  int bk = blockIdx.x;        // (b,k)
  int n = threadIdx.x;
  int b = bk >> 2;
  float x = score[(size_t)bk * 256 + n];
  sh[n] = x; __syncthreads();
  for (int s = 128; s > 0; s >>= 1) { if (n < s) sh[n] = fmaxf(sh[n], sh[n + s]); __syncthreads(); }
  float mx = sh[0]; __syncthreads();
  float e = __expf(x - mx) * vmask[b * 256 + n];
  sh[n] = e; __syncthreads();
  for (int s = 128; s > 0; s >>= 1) { if (n < s) sh[n] += sh[n + s]; __syncthreads(); }
  score[(size_t)bk * 256 + n] = e / (sh[0] + 1e-6f);
}

__global__ void k_m2s(const float* __restrict__ attn, const float* __restrict__ wmv,
                      float* __restrict__ m2s, bf16* __restrict__ m2sbf) {
  int bk = blockIdx.x;        // (b,k)
  int g = threadIdx.x;        // 128
  int b = bk >> 2, k = bk & 3;
  float acc = 0.f;
  for (int n = 0; n < 256; ++n)
    acc += attn[(size_t)bk * 256 + n] * wmv[((size_t)b * 256 + n) * 512 + k * 128 + g];
  int o = b * 512 + k * 128 + g;
  m2s[o] = acc;
  m2sbf[o] = (bf16)acc;
}

__global__ void k_gate(const float* __restrict__ z, const float* __restrict__ x,
                       const float* __restrict__ y, int shift,
                       float* __restrict__ out, bf16* __restrict__ outbf,
                       int ldob, int off, int M) {
  int i = (int)(blockIdx.x * blockDim.x + threadIdx.x);
  if (i >= M * 128) return;
  int r = i >> 7, c = i & 127;
  float zz = z[i];
  float h = (1.f - zz) * x[i] + zz * y[((r >> shift) << 7) + c];
  out[i] = h;
  outbf[(size_t)r * ldob + off + c] = (bf16)h;
}

__global__ void k_gru(const float* __restrict__ gi, const float* __restrict__ gh,
                      const float* __restrict__ hprev,
                      float* __restrict__ out, bf16* __restrict__ outbf,
                      int ldob, int off, int M) {
  int i = (int)(blockIdx.x * blockDim.x + threadIdx.x);
  if (i >= M * 128) return;
  int r = i >> 7, c = i & 127;
  size_t base = (size_t)r * 384;
  float ir = gi[base + c], iz = gi[base + 128 + c], inn = gi[base + 256 + c];
  float hr = gh[base + c], hz = gh[base + 128 + c], hn  = gh[base + 256 + c];
  float rr = sigmoidf_(ir + hr);
  float zz = sigmoidf_(iz + hz);
  float nn = tanhf(inn + rr * hn);
  float h = (1.f - zz) * nn + zz * hprev[i];
  out[i] = h;
  outbf[(size_t)r * ldob + off + c] = (bf16)h;
}

// ---------------------------------------------------------------------------
// Host driver
// ---------------------------------------------------------------------------
extern "C" void kernel_launch(void* const* d_in, const int* in_sizes, int n_in,
                              void* d_out, int out_size, void* d_ws, size_t ws_size,
                              hipStream_t stream) {
  (void)in_sizes; (void)n_in; (void)out_size; (void)ws_size;
  const int M = 65536, ME = 131072;     // B*N, B*E

  const float* vertex_mask = (const float*)d_in[0];
  const float* nbs_mask    = (const float*)d_in[1];
  const int*   vertex      = (const int*)d_in[2];
  const int*   edge        = (const int*)d_in[3];
  const int*   atom_adj    = (const int*)d_in[4];
  const int*   bond_adj    = (const int*)d_in[5];
  const float* af          = (const float*)d_in[6];
  const float* bfeat       = (const float*)d_in[7];
  const float* W_ve = (const float*)d_in[8];  const float* b_ve = (const float*)d_in[9];
  const float* Wa   = (const float*)d_in[10]; const float* ba   = (const float*)d_in[11];
  const float* Wm   = (const float*)d_in[12]; const float* bm   = (const float*)d_in[13];
  const float* Wbmm = (const float*)d_in[14]; const float* bbmm = (const float*)d_in[15];
  const float* Wm2s = (const float*)d_in[16]; const float* bm2s = (const float*)d_in[17];
  const float* Ws2m = (const float*)d_in[18]; const float* bs2m = (const float*)d_in[19];
  const float* Ws   = (const float*)d_in[20]; const float* bs   = (const float*)d_in[21];
  const float* Wzm1 = (const float*)d_in[22]; const float* bzm1 = (const float*)d_in[23];
  const float* Wzm2 = (const float*)d_in[24]; const float* bzm2 = (const float*)d_in[25];
  const float* Wzs1 = (const float*)d_in[26]; const float* bzs1 = (const float*)d_in[27];
  const float* Wzs2 = (const float*)d_in[28]; const float* bzs2 = (const float*)d_in[29];
  const float* Wu2  = (const float*)d_in[30]; const float* bu2  = (const float*)d_in[31];
  const float* Wu1  = (const float*)d_in[32]; const float* bu1  = (const float*)d_in[33];
  const float* gmwi = (const float*)d_in[34]; const float* gmwh = (const float*)d_in[35];
  const float* gmbi = (const float*)d_in[36]; const float* gmbh = (const float*)d_in[37];
  const float* gswi = (const float*)d_in[38]; const float* gswh = (const float*)d_in[39];
  const float* gsbi = (const float*)d_in[40]; const float* gsbh = (const float*)d_in[41];

  // ---- workspace bump allocator ----
  char* p = (char*)d_ws;
  auto alloc = [&](size_t bytes) -> char* {
    char* r = p; p += (bytes + 255) & ~(size_t)255; return r;
  };
  bf16* WveT  = (bf16*)alloc((size_t)128 * 96 * 2);
  bf16* WaT   = (bf16*)alloc((size_t)3 * 512 * 128 * 2);
  bf16* WmT   = (bf16*)alloc((size_t)3 * 512 * 128 * 2);
  bf16* Wm2sT = (bf16*)alloc((size_t)3 * 128 * 512 * 2);
  bf16* Ws2mT = (bf16*)alloc((size_t)3 * 128 * 128 * 2);
  bf16* WsT   = (bf16*)alloc((size_t)3 * 128 * 128 * 2);
  bf16* Wzm1T = (bf16*)alloc((size_t)3 * 128 * 128 * 2);
  bf16* Wzm2T = (bf16*)alloc((size_t)3 * 128 * 128 * 2);
  bf16* Wzs1T = (bf16*)alloc((size_t)3 * 128 * 128 * 2);
  bf16* Wzs2T = (bf16*)alloc((size_t)3 * 128 * 128 * 2);
  bf16* Wu2T  = (bf16*)alloc((size_t)3 * 128 * 160 * 2);
  bf16* Wu1T  = (bf16*)alloc((size_t)3 * 128 * 256 * 2);
  bf16* wim   = (bf16*)alloc((size_t)384 * 128 * 2);
  bf16* whm   = (bf16*)alloc((size_t)384 * 128 * 2);
  bf16* wis   = (bf16*)alloc((size_t)384 * 128 * 2);
  bf16* whs   = (bf16*)alloc((size_t)384 * 128 * 2);
  float* cbzm = (float*)alloc((size_t)3 * 128 * 4);
  float* cbzs = (float*)alloc((size_t)3 * 128 * 4);

  bf16* vg   = (bf16*)alloc((size_t)M * 96 * 2);
  bf16* ebf  = (bf16*)alloc((size_t)ME * 32 * 2);
  bf16* cat  = (bf16*)alloc((size_t)M * 256 * 2);   // [vf_bf16 | nei_bf16]
  float* vf  = (float*)alloc((size_t)M * 128 * 4);
  float* big1 = (float*)alloc((size_t)M * 512 * 4);  // a_main, then GRU gi
  float* big2 = (float*)alloc((size_t)M * 512 * 4);  // wm_v,  then GRU gh
  float* attn = (float*)alloc((size_t)256 * 4 * 256 * 4);
  float* m2s  = (float*)alloc((size_t)256 * 512 * 4);
  bf16*  m2sbf = (bf16*)alloc((size_t)256 * 512 * 2);
  float* mts  = (float*)alloc((size_t)256 * 128 * 4);
  bf16*  mtsbf = (bf16*)alloc((size_t)256 * 128 * 2);
  float* s2m  = (float*)alloc((size_t)256 * 128 * 4);
  bf16*  s2mbf = (bf16*)alloc((size_t)256 * 128 * 2);
  float* ssb  = (float*)alloc((size_t)256 * 128 * 4);
  bf16*  ssbf = (bf16*)alloc((size_t)256 * 128 * 2);
  float* t2   = (float*)alloc((size_t)256 * 128 * 4);
  float* zsup = (float*)alloc((size_t)256 * 128 * 4);
  float* hs   = (float*)alloc((size_t)256 * 128 * 4);
  bf16*  hsbf = (bf16*)alloc((size_t)256 * 128 * 2);
  float* gis  = (float*)alloc((size_t)256 * 384 * 4);
  float* ghs  = (float*)alloc((size_t)256 * 384 * 4);
  float* sf   = (float*)alloc((size_t)256 * 128 * 4);
  bf16*  sfbf = (bf16*)alloc((size_t)256 * 128 * 2);
  float* nei  = (float*)alloc((size_t)M * 128 * 4);
  float* mself = (float*)alloc((size_t)M * 128 * 4);
  bf16*  msbf  = (bf16*)alloc((size_t)M * 128 * 2);
  float* zbuf  = (float*)alloc((size_t)M * 128 * 4);
  float* hid   = (float*)alloc((size_t)M * 128 * 4);
  bf16*  hidbf = (bf16*)alloc((size_t)M * 128 * 2);

  auto gemm = [&](const bf16* A, int lda, const bf16* Bt, int ldb,
                  float* C, bf16* Cbf, int ldcb, int off,
                  const float* bias, const float* rb, int rbs,
                  int Mm, int Nt, int K, int epi) {
    int waves = (Mm >> 4) * (Nt >> 6);
    int blocks = (waves + 7) >> 3;
    k_gemm<<<blocks, 256, 0, stream>>>(A, lda, Bt, ldb, C, Cbf, ldcb, off,
                                       bias, rb, rbs, Mm, Nt, K, epi);
  };
  auto tpose = [&](const float* in, bf16* out, int Kd, int Nn, int ldo) {
    int tot = Nn * ldo;
    k_transpose_pad<<<(tot + 255) / 256, 256, 0, stream>>>(in, out, Kd, Nn, ldo);
  };

  // ---- weight prep (bf16, B^T row-major) ----
  tpose(W_ve, WveT, 82, 128, 96);
  for (int d = 0; d < 3; ++d) {
    for (int k = 0; k < 4; ++k) {
      tpose(Wa + (size_t)(d * 4 + k) * 16384, WaT + ((size_t)d * 512 + k * 128) * 128, 128, 128, 128);
      tpose(Wm + (size_t)(d * 4 + k) * 16384, WmT + ((size_t)d * 512 + k * 128) * 128, 128, 128, 128);
    }
    tpose(Wm2s + (size_t)d * 512 * 128, Wm2sT + (size_t)d * 128 * 512, 512, 128, 512);
    tpose(Ws2m + (size_t)d * 16384, Ws2mT + (size_t)d * 16384, 128, 128, 128);
    tpose(Ws   + (size_t)d * 16384, WsT   + (size_t)d * 16384, 128, 128, 128);
    tpose(Wzm1 + (size_t)d * 16384, Wzm1T + (size_t)d * 16384, 128, 128, 128);
    tpose(Wzm2 + (size_t)d * 16384, Wzm2T + (size_t)d * 16384, 128, 128, 128);
    tpose(Wzs1 + (size_t)d * 16384, Wzs1T + (size_t)d * 16384, 128, 128, 128);
    tpose(Wzs2 + (size_t)d * 16384, Wzs2T + (size_t)d * 16384, 128, 128, 128);
    tpose(Wu2 + (size_t)d * 134 * 128, Wu2T + (size_t)d * 128 * 160, 134, 128, 160);
    tpose(Wu1 + (size_t)d * 256 * 128, Wu1T + (size_t)d * 128 * 256, 256, 128, 256);
  }
  // GRU weights: x@wi.T means B^T = wi itself (row = output col) -> plain convert.
  k_copy_bf16<<<192, 256, 0, stream>>>(gmwi, wim, 384 * 128);
  k_copy_bf16<<<192, 256, 0, stream>>>(gmwh, whm, 384 * 128);
  k_copy_bf16<<<192, 256, 0, stream>>>(gswi, wis, 384 * 128);
  k_copy_bf16<<<192, 256, 0, stream>>>(gswh, whs, 384 * 128);
  k_addvec<<<2, 256, 0, stream>>>(bzm1, bzm2, cbzm, 384);
  k_addvec<<<2, 256, 0, stream>>>(bzs1, bzs2, cbzs, 384);

  // ---- embeddings + initial vertex features / super node ----
  k_gather_pad<<<((M * 96) + 255) / 256, 256, 0, stream>>>(af, 82, vertex, vg, 96, 82, M);
  k_gather_pad<<<((ME * 32) + 255) / 256, 256, 0, stream>>>(bfeat, 6, edge, ebf, 32, 6, ME);
  gemm(vg, 96, WveT, 96, vf, cat, 256, 0, b_ve, nullptr, 0, M, 128, 96, 1 /*lrelu*/);
  k_sf_init<<<256, 128, 0, stream>>>(vf, vertex_mask, sf, sfbf);

  // ---- depth loop ----
  for (int d = 0; d < 3; ++d) {
    const bf16* WaT_d   = WaT   + (size_t)d * 512 * 128;
    const bf16* WmT_d   = WmT   + (size_t)d * 512 * 128;
    const bf16* Wm2sT_d = Wm2sT + (size_t)d * 128 * 512;
    const bf16* Ws2mT_d = Ws2mT + (size_t)d * 16384;
    const bf16* WsT_d   = WsT   + (size_t)d * 16384;
    const bf16* Wzm1T_d = Wzm1T + (size_t)d * 16384;
    const bf16* Wzm2T_d = Wzm2T + (size_t)d * 16384;
    const bf16* Wzs1T_d = Wzs1T + (size_t)d * 16384;
    const bf16* Wzs2T_d = Wzs2T + (size_t)d * 16384;
    const bf16* Wu2T_d  = Wu2T  + (size_t)d * 128 * 160;
    const bf16* Wu1T_d  = Wu1T  + (size_t)d * 128 * 256;

    // multi-head attention: a_main, wm_v
    gemm(cat, 256, WaT_d, 128, big1, nullptr, 0, 0, ba + d * 512, nullptr, 0, M, 512, 128, 2 /*tanh*/);
    gemm(cat, 256, WmT_d, 128, big2, nullptr, 0, 0, bm + d * 512, nullptr, 0, M, 512, 128, 0);
    k_attn_score<<<1024, 256, 0, stream>>>(big1, sf, Wbmm + d * 512, bbmm + d * 4, attn);
    k_softmax<<<1024, 256, 0, stream>>>(attn, vertex_mask);
    k_m2s<<<1024, 128, 0, stream>>>(attn, big2, m2s, m2sbf);
    gemm(m2sbf, 512, Wm2sT_d, 512, mts, mtsbf, 128, 0, bm2s + d * 128, nullptr, 0, 256, 128, 512, 2);

    // WLN neighbor aggregation (fused gather + GEMM + lrelu + masked NB-sum)
    k_nei<<<16384, 256, 0, stream>>>(cat, ebf, atom_adj, bond_adj, Wu2T_d,
                                     bu2 + d * 128, nbs_mask, nei, cat);
    gemm(cat, 256, Wu1T_d, 256, mself, msbf, 128, 0, bu1 + d * 128, nullptr, 0, M, 128, 256, 1);

    // gated fusion (main)
    gemm(sfbf, 128, Ws2mT_d, 128, s2m, s2mbf, 128, 0, bs2m + d * 128, nullptr, 0, 256, 128, 128, 2);
    gemm(s2mbf, 128, Wzm2T_d, 128, t2, nullptr, 0, 0, cbzm + d * 128, nullptr, 0, 256, 128, 128, 0);
    gemm(msbf, 128, Wzm1T_d, 128, zbuf, nullptr, 0, 0, nullptr, t2, 8, M, 128, 128, 3 /*sigmoid*/);
    k_gate<<<((M * 128) + 255) / 256, 256, 0, stream>>>(zbuf, mself, s2m, 8, hid, hidbf, 128, 0, M);

    // GRU (main)
    gemm(hidbf, 128, wim, 128, big1, nullptr, 0, 0, gmbi, nullptr, 0, M, 384, 128, 0);
    gemm(cat, 256, whm, 128, big2, nullptr, 0, 0, gmbh, nullptr, 0, M, 384, 128, 0);  // old vf
    float* vfdst = (d == 2) ? (float*)d_out : vf;
    k_gru<<<((M * 128) + 255) / 256, 256, 0, stream>>>(big1, big2, vf, vfdst, cat, 256, 0, M);

    // gated fusion + GRU (super node)
    gemm(sfbf, 128, WsT_d, 128, ssb, ssbf, 128, 0, bs + d * 128, nullptr, 0, 256, 128, 128, 2);
    gemm(mtsbf, 128, Wzs1T_d, 128, t2, nullptr, 0, 0, nullptr, nullptr, 0, 256, 128, 128, 0);
    gemm(ssbf, 128, Wzs2T_d, 128, zsup, nullptr, 0, 0, cbzs + d * 128, t2, 0, 256, 128, 128, 3);
    k_gate<<<128, 256, 0, stream>>>(zsup, mts, ssb, 0, hs, hsbf, 128, 0, 256);
    gemm(hsbf, 128, wis, 128, gis, nullptr, 0, 0, gsbi, nullptr, 0, 256, 384, 128, 0);
    gemm(sfbf, 128, whs, 128, ghs, nullptr, 0, 0, gsbh, nullptr, 0, 256, 384, 128, 0);
    k_gru<<<128, 256, 0, stream>>>(gis, ghs, sf, sf, sfbf, 128, 0, 256);
  }
}
